// FlashSantacoderModel_68822555951402
// MI455X (gfx1250) — matrix operand order repaired and checked by reading, not compiled, not run on hardware
//
#include <hip/hip_runtime.h>
#include <hip/hip_bf16.h>

// ---------------------------------------------------------------------------
// Santacoder MQA prefill on gfx1250 (MI455X): f16 WMMA GEMMs + flash attention
// Round 3: 64x64 wave tiles (block 128x256) -> WMMA:DS issue ratio 1:1,
// double-buffered mainloop, transposed weights, ds_swizzle softmax.
// ---------------------------------------------------------------------------

typedef __attribute__((ext_vector_type(16))) _Float16 v16h;
typedef __attribute__((ext_vector_type(8)))  float    v8f;

#define WMMA(a, b, c) __builtin_amdgcn_wmma_f32_16x16x32_f16( \
    false, (a), false, (b), (short)0, (c), false, false)

// ds_swizzle group-of-32 XOR butterfly: lane <- lane ^ m (single DS op).
#define SWZ_XOR(x, m) \
  __int_as_float(__builtin_amdgcn_ds_swizzle(__float_as_int(x), ((m) << 10) | 0x1f))

constexpr int Bc  = 2;
constexpr int Sc  = 1024;
constexpr int Dc  = 2048;
constexpr int Hc  = 16;
constexpr int HDc = 128;
constexpr int FFc = 8192;
constexpr int Lc  = 2;
constexpr int Tc  = Bc * Sc;          // 2048 tokens
constexpr int QKVN = Dc + 2 * HDc;    // 2304
constexpr float EPSc   = 1e-5f;
constexpr float SCALEc = 0.08838834764831845f;  // 128^-0.5
constexpr float NEGINF = -3.0e38f;

// ---------------------------------------------------------------------------
// WMMA fragment loaders (CDNA5 16x16x32 f16 layouts, ISA 7.12.2)
// ---------------------------------------------------------------------------

// A: 16x32 (MxK), row-major source, leading dim `ld` (halves). Per lane the
// 16 halves are two contiguous 8-half runs -> 2x ds_load_b128.
__device__ inline v16h load_a_frag(const _Float16* __restrict__ base, int ld) {
  int lane = threadIdx.x & 31;
  const _Float16* p = base + (lane & 15) * ld + ((lane >> 4) << 3);
  v16h a;
#pragma unroll
  for (int d = 0; d < 8; ++d) {
    int k = ((d >> 2) << 4) + ((d & 3) << 1);
    a[2 * d]     = p[k];
    a[2 * d + 1] = p[k + 1];
  }
  return a;
}

// B from an N-major (transposed) source: element (k,n) = base[n*ld + k].
// Per lane: two contiguous 8-half runs (same k order as A) -> 2x ds_load_b128.
__device__ inline v16h load_bT_frag(const _Float16* __restrict__ base, int ld) {
  int lane = threadIdx.x & 31;
  const _Float16* p = base + (lane & 15) * ld + ((lane >> 4) << 3);
  v16h b;
#pragma unroll
  for (int d = 0; d < 8; ++d) {
    int k = ((d >> 2) << 4) + ((d & 3) << 1);
    b[2 * d]     = p[k];
    b[2 * d + 1] = p[k + 1];
  }
  return b;
}

// ---------------------------------------------------------------------------
// Setup kernels
// ---------------------------------------------------------------------------

// Transpose-convert: in [K,N] f32 -> out [N,K] f16, 32x32 LDS tiles, both
// sides coalesced. grid = (N/32, K/32, L).
__global__ __launch_bounds__(256) void cvtT_kernel(
    const float* __restrict__ in, _Float16* __restrict__ outp, int K, int N) {
  __shared__ float tile[32][33];
  in   += (size_t)blockIdx.z * K * N;
  outp += (size_t)blockIdx.z * K * N;
  int k0 = blockIdx.y * 32, n0 = blockIdx.x * 32;
  int tx = threadIdx.x & 31, ty = threadIdx.x >> 5;  // 32 x 8
#pragma unroll
  for (int i = 0; i < 32; i += 8)
    tile[ty + i][tx] = in[(size_t)(k0 + ty + i) * N + n0 + tx];
  __syncthreads();
#pragma unroll
  for (int i = 0; i < 32; i += 8)
    outp[(size_t)(n0 + ty + i) * K + k0 + tx] = (_Float16)tile[tx][ty + i];
}

__global__ void embed_kernel(const int* __restrict__ ids,
                             const int* __restrict__ pos,
                             const float* __restrict__ wte,
                             const float* __restrict__ wpe,
                             float* __restrict__ h) {
  int t = blockIdx.x;
  size_t wo = (size_t)ids[t] * Dc;
  size_t po = (size_t)pos[t] * Dc;
  for (int d = threadIdx.x; d < Dc; d += blockDim.x)
    h[(size_t)t * Dc + d] = wte[wo + d] + wpe[po + d];
}

// Fused residual-add + LayerNorm. Optionally writes: pre-LN sum (res_out),
// f16 normalized activations (x16), f32 normalized output (x32).
__global__ __launch_bounds__(256) void ln_kernel(
    const float* __restrict__ in, const float* __restrict__ resid,
    const float* __restrict__ g, const float* __restrict__ bta,
    float* __restrict__ res_out, _Float16* __restrict__ x16,
    float* __restrict__ x32) {
  int t = blockIdx.x;
  const float* pi = in + (size_t)t * Dc;
  const float* pr = resid ? resid + (size_t)t * Dc : nullptr;
  float v[8];
  float s = 0.f, ss = 0.f;
#pragma unroll
  for (int i = 0; i < 8; ++i) {
    int d = threadIdx.x + i * 256;
    float x = pi[d] + (pr ? pr[d] : 0.f);
    v[i] = x;
    s += x;
    ss += x * x;
  }
  // wave32 butterfly via ds_swizzle, then cross-wave via LDS
  s += SWZ_XOR(s, 1);  ss += SWZ_XOR(ss, 1);
  s += SWZ_XOR(s, 2);  ss += SWZ_XOR(ss, 2);
  s += SWZ_XOR(s, 4);  ss += SWZ_XOR(ss, 4);
  s += SWZ_XOR(s, 8);  ss += SWZ_XOR(ss, 8);
  s += SWZ_XOR(s, 16); ss += SWZ_XOR(ss, 16);
  __shared__ float red[16];
  int lane = threadIdx.x & 31, wid = threadIdx.x >> 5;
  if (lane == 0) { red[wid] = s; red[8 + wid] = ss; }
  __syncthreads();
  s = 0.f; ss = 0.f;
#pragma unroll
  for (int w = 0; w < 8; ++w) { s += red[w]; ss += red[8 + w]; }
  float mu   = s * (1.f / Dc);
  float var  = ss * (1.f / Dc) - mu * mu;
  float rstd = rsqrtf(var + EPSc);
#pragma unroll
  for (int i = 0; i < 8; ++i) {
    int d = threadIdx.x + i * 256;
    if (res_out) res_out[(size_t)t * Dc + d] = v[i];
    float y = (v[i] - mu) * rstd * g[d] + bta[d];
    if (x16) x16[(size_t)t * Dc + d] = (_Float16)y;
    if (x32) x32[(size_t)t * Dc + d] = y;
  }
}

// ---------------------------------------------------------------------------
// WMMA GEMM: C[M,N] = A[M,K](f16) x BT[N,K](f16, transposed) + bias.
//   EPI 0: +bias -> f16, with V columns scattered to vT[B,HD,S]   (QKV)
//   EPI 1: +bias +residual  -> f32                                 (attn proj)
//   EPI 2: +bias, tanh-GELU -> f16                                 (fc)
//   EPI 3: +bias            -> f32                                 (mlp proj)
// Block: 256 thr (8 waves, 2x4), tile 128(M) x 256(N), K-step 64, double-
// buffered. Wave tile 64x64: 16 WMMA per 16 ds_load_b128 per k-step.
// All dims are multiples of the tile sizes for this model.
// ---------------------------------------------------------------------------

constexpr int BMt = 128, BNt = 256, BKt = 64;
constexpr int ALD = 80;  // 64 + 16 halves pad (16B-aligned rows)

__device__ inline float gelu_tanh(float x) {
  float x3 = x * x * x;
  return 0.5f * x * (1.0f + tanhf(0.7978845608028654f * (x + 0.044715f * x3)));
}

template <int EPI>
__global__ __launch_bounds__(256) void gemm_kernel(
    const _Float16* __restrict__ A, const _Float16* __restrict__ BT,
    const float* __restrict__ bias, const float* __restrict__ resid,
    void* __restrict__ outv, _Float16* __restrict__ vT,
    int M, int N, int K) {
  __shared__ __align__(16) _Float16 As[2][BMt * ALD];  // [m][k]
  __shared__ __align__(16) _Float16 Bs[2][BNt * ALD];  // [n][k], n-major

  int tid  = threadIdx.x;
  int lane = tid & 31;
  int wid  = tid >> 5;
  int wm   = wid & 1;        // 2 wave-rows of 64 (M)
  int wn   = wid >> 1;       // 4 wave-cols of 64 (N)
  int m0   = blockIdx.y * BMt;
  int n0   = blockIdx.x * BNt;

  v8f vz = {};
  v8f acc[4][4];
#pragma unroll
  for (int i = 0; i < 4; ++i)
#pragma unroll
    for (int j = 0; j < 4; ++j) acc[i][j] = vz;

  uint4 ra[4], rb[8];
  auto gload = [&](int kt) {
#pragma unroll
    for (int i = 0; i < 4; ++i) {
      int idx = tid + i * 256;
      int r = idx >> 3, c = (idx & 7) * 8;
      ra[i] = *(const uint4*)&A[(size_t)(m0 + r) * K + kt + c];
    }
#pragma unroll
    for (int i = 0; i < 8; ++i) {
      int idx = tid + i * 256;
      int r = idx >> 3, c = (idx & 7) * 8;
      rb[i] = *(const uint4*)&BT[(size_t)(n0 + r) * K + kt + c];
    }
  };
  auto sstore = [&](int buf) {
#pragma unroll
    for (int i = 0; i < 4; ++i) {
      int idx = tid + i * 256;
      int r = idx >> 3, c = (idx & 7) * 8;
      *(uint4*)&As[buf][r * ALD + c] = ra[i];
    }
#pragma unroll
    for (int i = 0; i < 8; ++i) {
      int idx = tid + i * 256;
      int r = idx >> 3, c = (idx & 7) * 8;
      *(uint4*)&Bs[buf][r * ALD + c] = rb[i];
    }
  };

  gload(0);
  sstore(0);
  __syncthreads();

  int nk = K / BKt;
  for (int t = 0; t < nk; ++t) {
    int cur = t & 1;
    if (t + 1 < nk) {
      gload((t + 1) * BKt);  // VMEM latency overlapped with WMMA below
      if (t + 2 < nk)        // speculative L2 prefetch of tile t+2
        __builtin_prefetch(&A[(size_t)(m0 + (tid >> 3)) * K + (t + 2) * BKt], 0, 0);
    }
#pragma unroll
    for (int kk = 0; kk < BKt; kk += 32) {
      v16h af[4];
#pragma unroll
      for (int i = 0; i < 4; ++i)
        af[i] = load_a_frag(&As[cur][(wm * 64 + i * 16) * ALD + kk], ALD);
#pragma unroll
      for (int j = 0; j < 4; ++j) {
        v16h bf = load_bT_frag(&Bs[cur][(wn * 64 + j * 16) * ALD + kk], ALD);
#pragma unroll
        for (int i = 0; i < 4; ++i)
          acc[i][j] = WMMA(af[i], bf, acc[i][j]);
      }
    }
    if (t + 1 < nk) {
      __syncthreads();       // everyone finished reading buf cur^1
      sstore(cur ^ 1);
      __syncthreads();       // stores visible before next compute
    }
  }

  // Epilogue. C layout: lane = N (mod 16); VGPR r = row r (+8 upper half).
#pragma unroll
  for (int i = 0; i < 4; ++i) {
#pragma unroll
    for (int j = 0; j < 4; ++j) {
#pragma unroll
      for (int r = 0; r < 8; ++r) {
        int gm = m0 + wm * 64 + i * 16 + r + ((lane >> 4) << 3);
        int gn = n0 + wn * 64 + j * 16 + (lane & 15);
        float val = acc[i][j][r] + bias[gn];
        if (EPI == 1) val += resid[(size_t)gm * N + gn];
        if (EPI == 2) val = gelu_tanh(val);
        if (EPI == 0) {
          if (gn < Dc + HDc) {
            ((_Float16*)outv)[(size_t)gm * N + gn] = (_Float16)val;
          } else {  // V head -> transposed [B, HD, S] for the PV WMMA
            int bb = gm >> 10, s = gm & (Sc - 1);
            vT[((size_t)bb * HDc + (gn - (Dc + HDc))) * Sc + s] = (_Float16)val;
          }
        } else if (EPI == 2) {
          ((_Float16*)outv)[(size_t)gm * N + gn] = (_Float16)val;
        } else {
          ((float*)outv)[(size_t)gm * N + gn] = val;
        }
      }
    }
  }
}

// ---------------------------------------------------------------------------
// Flash-style MQA attention (single shared KV head).
// Block = (q-tile of 64, head, batch); 128 threads = 4 waves, wave owns 16 q
// rows. K streamed row-major (QK^T contracts over contiguous HD); V streamed
// from the transposed vT buffer so PV fragments are also contiguous.
// ---------------------------------------------------------------------------

constexpr int KLD  = 136;  // 128 + 8 halves pad (272B rows)
constexpr int VTLD = 40;   // 32 + 8 halves pad (80B rows)
constexpr int PLD  = 40;

__global__ __launch_bounds__(128) void attn_kernel(
    const _Float16* __restrict__ qkv, const _Float16* __restrict__ vT,
    _Float16* __restrict__ ctx) {
  __shared__ __align__(16) _Float16 Kt[32 * KLD];     // [key][hd]
  __shared__ __align__(16) _Float16 Vt[HDc * VTLD];   // [hd][key]
  __shared__ __align__(16) _Float16 Pl[4 * 16 * PLD];

  int tid   = threadIdx.x;
  int lane  = tid & 31;
  int wid   = tid >> 5;
  int qtile = blockIdx.x;
  int head  = blockIdx.y;
  int b     = blockIdx.z;
  int qrow0 = qtile * 64 + wid * 16;

  // Q fragments (16 rows x 128 HD), loaded once from global.
  v16h qf[4];
#pragma unroll
  for (int kc = 0; kc < 4; ++kc)
    qf[kc] = load_a_frag(
        qkv + (size_t)(b * Sc + qrow0) * QKVN + head * HDc + kc * 32, QKVN);

  v8f vz = {};
  v8f o[8];
#pragma unroll
  for (int fn = 0; fn < 8; ++fn) o[fn] = vz;
  float mrow[8], lrow[8];
#pragma unroll
  for (int r = 0; r < 8; ++r) { mrow[r] = NEGINF; lrow[r] = 0.f; }

  int kend = qtile * 64 + 64;
  for (int kb = 0; kb < kend; kb += 32) {
    // K tile: 32 keys x 128 HD; V tile: 128 HD x 32 keys (both b128 stores).
#pragma unroll
    for (int i = 0; i < 4; ++i) {
      int idx = tid + i * 128;
      { int r = idx >> 4, c = (idx & 15) * 8;
        *(uint4*)&Kt[r * KLD + c] =
            *(const uint4*)&qkv[(size_t)(b * Sc + kb + r) * QKVN + Dc + c]; }
      { int r = idx >> 2, c = (idx & 3) * 8;
        *(uint4*)&Vt[r * VTLD + c] =
            *(const uint4*)&vT[((size_t)b * HDc + r) * Sc + kb + c]; }
    }
    __syncthreads();

    if (kb < qrow0 + 16) {  // wave-uniform: tile intersects causal region
      // S = Q * K^T  (two 16x16 key fragments, contraction over HD=128)
      v8f sc0 = vz, sc1 = vz;
#pragma unroll
      for (int kc = 0; kc < 4; ++kc) {
        v16h bk0 = load_bT_frag(&Kt[0 * KLD + kc * 32], KLD);
        v16h bk1 = load_bT_frag(&Kt[16 * KLD + kc * 32], KLD);
        sc0 = WMMA(qf[kc], bk0, sc0);
        sc1 = WMMA(qf[kc], bk1, sc1);
      }

      // Online softmax per row (rows split across lane halves).
      int colg = kb + (lane & 15);
#pragma unroll
      for (int r = 0; r < 8; ++r) {
        int qi = qrow0 + r + ((lane >> 4) << 3);
        float s0 = sc0[r] * SCALEc; if (colg > qi)      s0 = NEGINF;
        float s1 = sc1[r] * SCALEc; if (colg + 16 > qi) s1 = NEGINF;
        float mx = fmaxf(s0, s1);
        mx = fmaxf(mx, SWZ_XOR(mx, 1));
        mx = fmaxf(mx, SWZ_XOR(mx, 2));
        mx = fmaxf(mx, SWZ_XOR(mx, 4));
        mx = fmaxf(mx, SWZ_XOR(mx, 8));
        float mnew = fmaxf(mrow[r], mx);
        float p0 = __expf(s0 - mnew);
        float p1 = __expf(s1 - mnew);
        float rs = p0 + p1;
        rs += SWZ_XOR(rs, 1);
        rs += SWZ_XOR(rs, 2);
        rs += SWZ_XOR(rs, 4);
        rs += SWZ_XOR(rs, 8);
        float fac = __expf(mrow[r] - mnew);
        lrow[r] = lrow[r] * fac + rs;
        mrow[r] = mnew;
#pragma unroll
        for (int fn = 0; fn < 8; ++fn) o[fn][r] *= fac;
        int prow = r + ((lane >> 4) << 3);
        Pl[(wid * 16 + prow) * PLD + (lane & 15)]      = (_Float16)p0;
        Pl[(wid * 16 + prow) * PLD + 16 + (lane & 15)] = (_Float16)p1;
      }
      // Same-wave LDS RAW: P written above, re-read as A-fragment below.
      asm volatile("s_wait_dscnt 0" ::: "memory");

      // O += P(16x32) * V(32x128); V fragments contiguous via Vt [hd][key].
      v16h ap = load_a_frag(&Pl[wid * 16 * PLD], PLD);
#pragma unroll
      for (int fn = 0; fn < 8; ++fn) {
        v16h bv = load_bT_frag(&Vt[(fn * 16) * VTLD], VTLD);
        o[fn] = WMMA(ap, bv, o[fn]);
      }
    }
    __syncthreads();  // protect K/V tiles before next iteration's overwrite
  }

  // Normalize and write ctx as f16 for the proj GEMM.
#pragma unroll
  for (int fn = 0; fn < 8; ++fn) {
#pragma unroll
    for (int r = 0; r < 8; ++r) {
      int qi = qrow0 + r + ((lane >> 4) << 3);
      float val = o[fn][r] / lrow[r];
      ctx[(size_t)(b * Sc + qi) * Dc + head * HDc + fn * 16 + (lane & 15)] =
          (_Float16)val;
    }
  }
}

// ---------------------------------------------------------------------------
// Host orchestration
// ---------------------------------------------------------------------------

extern "C" void kernel_launch(void* const* d_in, const int* in_sizes, int n_in,
                              void* d_out, int out_size, void* d_ws,
                              size_t ws_size, hipStream_t stream) {
  (void)in_sizes; (void)n_in; (void)out_size; (void)ws_size;
  const int*   ids  = (const int*)d_in[0];
  const int*   pos  = (const int*)d_in[1];
  const float* wte  = (const float*)d_in[2];
  const float* wpe  = (const float*)d_in[3];
  const float* caw  = (const float*)d_in[4];
  const float* cab  = (const float*)d_in[5];
  const float* cpw  = (const float*)d_in[6];
  const float* cpb  = (const float*)d_in[7];
  const float* l1w  = (const float*)d_in[8];
  const float* l1b  = (const float*)d_in[9];
  const float* l2w  = (const float*)d_in[10];
  const float* l2b  = (const float*)d_in[11];
  const float* fcw  = (const float*)d_in[12];
  const float* fcb  = (const float*)d_in[13];
  const float* mpw  = (const float*)d_in[14];
  const float* mpb  = (const float*)d_in[15];
  const float* lnfw = (const float*)d_in[16];
  const float* lnfb = (const float*)d_in[17];
  float* out = (float*)d_out;

  char* ws = (char*)d_ws;
  size_t off = 0;
  auto alloc = [&](size_t bytes) -> char* {
    char* p = ws + off;
    off = (off + bytes + 255) & ~(size_t)255;
    return p;
  };

  const size_t n_wq = (size_t)Lc * Dc * QKVN;
  const size_t n_wp = (size_t)Lc * Dc * Dc;
  const size_t n_wf = (size_t)Lc * Dc * FFc;
  const size_t n_wm = (size_t)Lc * FFc * Dc;

  _Float16* wq16  = (_Float16*)alloc(n_wq * 2);  // [L][QKVN][D]  (transposed)
  _Float16* wp16  = (_Float16*)alloc(n_wp * 2);  // [L][D][D]
  _Float16* wf16  = (_Float16*)alloc(n_wf * 2);  // [L][FF][D]
  _Float16* wm16  = (_Float16*)alloc(n_wm * 2);  // [L][D][FF]
  _Float16* x16   = (_Float16*)alloc((size_t)Tc * Dc * 2);
  _Float16* qkv16 = (_Float16*)alloc((size_t)Tc * QKVN * 2);
  _Float16* vT16  = (_Float16*)alloc((size_t)Bc * HDc * Sc * 2);
  _Float16* ctx16 = (_Float16*)alloc((size_t)Tc * Dc * 2);
  _Float16* fc16  = (_Float16*)alloc((size_t)Tc * FFc * 2);
  float*    hbuf  = (float*)alloc((size_t)Tc * Dc * 4);
  float*    resA  = (float*)alloc((size_t)Tc * Dc * 4);
  float*    resB  = (float*)alloc((size_t)Tc * Dc * 4);

  // One-time weight transpose + f16 downconversion (coalesced both sides).
  cvtT_kernel<<<dim3(QKVN / 32, Dc / 32, Lc), 256, 0, stream>>>(caw, wq16, Dc, QKVN);
  cvtT_kernel<<<dim3(Dc / 32,   Dc / 32, Lc), 256, 0, stream>>>(cpw, wp16, Dc, Dc);
  cvtT_kernel<<<dim3(FFc / 32,  Dc / 32, Lc), 256, 0, stream>>>(fcw, wf16, Dc, FFc);
  cvtT_kernel<<<dim3(Dc / 32,  FFc / 32, Lc), 256, 0, stream>>>(mpw, wm16, FFc, Dc);

  embed_kernel<<<Tc, 256, 0, stream>>>(ids, pos, wte, wpe, hbuf);

  dim3 gQKV(QKVN / BNt, Tc / BMt);  //  9 x 16
  dim3 gPRJ(Dc / BNt,   Tc / BMt);  //  8 x 16
  dim3 gFC (FFc / BNt,  Tc / BMt);  // 32 x 16
  dim3 gAT (Sc / 64, Hc, Bc);       // 16 x 16 x 2

  for (int l = 0; l < Lc; ++l) {
    const float* residual = (l == 0) ? nullptr : resB;
    // LN1: res = h (+ residual); x16 = LN(res)
    ln_kernel<<<Tc, 256, 0, stream>>>(hbuf, residual, l1w + (size_t)l * Dc,
                                      l1b + (size_t)l * Dc, resA, x16, nullptr);
    // QKV = x @ Wattn + b -> q,k into qkv16 (f16), v into vT16 [B,HD,S]
    gemm_kernel<0><<<gQKV, 256, 0, stream>>>(
        x16, wq16 + (size_t)l * Dc * QKVN, cab + (size_t)l * QKVN, nullptr,
        qkv16, vT16, Tc, QKVN, Dc);
    // Flash MQA attention -> ctx f16
    attn_kernel<<<gAT, 128, 0, stream>>>(qkv16, vT16, ctx16);
    // res2 = ctx @ Wproj + b + res -> f32
    gemm_kernel<1><<<gPRJ, 256, 0, stream>>>(
        ctx16, wp16 + (size_t)l * Dc * Dc, cpb + (size_t)l * Dc, resA,
        resB, nullptr, Tc, Dc, Dc);
    // LN2: x16 = LN(res2)
    ln_kernel<<<Tc, 256, 0, stream>>>(resB, nullptr, l2w + (size_t)l * Dc,
                                      l2b + (size_t)l * Dc, nullptr, x16,
                                      nullptr);
    // fcact = gelu(x2 @ Wfc + b) -> f16
    gemm_kernel<2><<<gFC, 256, 0, stream>>>(
        x16, wf16 + (size_t)l * Dc * FFc, fcb + (size_t)l * FFc, nullptr,
        fc16, nullptr, Tc, FFc, Dc);
    // h = fcact @ Wmp + b -> f32  (residual stays in resB)
    gemm_kernel<3><<<gPRJ, 256, 0, stream>>>(
        fc16, wm16 + (size_t)l * FFc * Dc, mpb + (size_t)l * Dc, nullptr,
        hbuf, nullptr, Tc, Dc, FFc);
  }

  // Final LN: out = LN(h + residual) -> f32 d_out
  ln_kernel<<<Tc, 256, 0, stream>>>(hbuf, resB, lnfw, lnfb, nullptr, nullptr,
                                    out);
}
